// SSA_SL_38173669327014
// MI455X (gfx1250) — 8-bit matrix operand order repaired and checked by reading, compiled, not run on hardware
//
#include <hip/hip_runtime.h>
#include <stdint.h>


typedef __attribute__((ext_vector_type(8))) int v8i;

#define T_  4
#define B_  16
#define NT_ 196
#define C_  384
#define H_  12
#define D_  32
#define M_  (T_ * B_ * NT_)
#define NPAD 208
#define MPAD 256
#define TN13 13

union AV { v8i v; long long d[4]; };
union BV { v8i v; int4 q[2]; };

__device__ __forceinline__ float cleanv(float v) {
    if (v != v) return 0.0f;
    if (fabsf(v) == __builtin_inff()) return 1.0f;
    return v;
}

__device__ __forceinline__ float pow2_scale(float m) {
    m = (m == 0.0f) ? 1.0f : m;
    return exp2f(floorf(log2f(floorf(32.0f / m))));
}


__global__ void absmax_kernel(const float* __restrict__ x, long long n,
                              float* __restrict__ out) {
    __shared__ float sh[256];
    float m = 0.0f;
    for (long long i = (long long)blockIdx.x * blockDim.x + threadIdx.x; i < n;
         i += (long long)gridDim.x * blockDim.x)
        m = fmaxf(m, fabsf(cleanv(x[i])));
    sh[threadIdx.x] = m;
    __syncthreads();
    for (int s = 128; s > 0; s >>= 1) {
        if ((int)threadIdx.x < s)
            sh[threadIdx.x] = fmaxf(sh[threadIdx.x], sh[threadIdx.x + s]);
        __syncthreads();
    }
    if (threadIdx.x == 0) atomicMax((int*)out, __float_as_int(sh[0]));
}

__global__ void quant_i8_kernel(const float* __restrict__ x,
                                const float* __restrict__ sc,
                                int8_t* __restrict__ out, long long n) {
    float SN = pow2_scale(*sc);
    for (long long i = (long long)blockIdx.x * blockDim.x + threadIdx.x; i < n;
         i += (long long)gridDim.x * blockDim.x)
        out[i] = (int8_t)(int)truncf(cleanv(x[i]) * SN);
}

__global__ void quant_f32_kernel(const float* __restrict__ x,
                                 const float* __restrict__ sc,
                                 float* __restrict__ out, int n) {
    float SN = pow2_scale(*sc);
    int i = blockIdx.x * blockDim.x + threadIdx.x;
    if (i < n) out[i] = truncf(cleanv(x[i]) * SN);
}


__global__ __launch_bounds__(256) void gemm_iu8_kernel(
    const int8_t* __restrict__ A, const int8_t* __restrict__ B,
    const float* __restrict__ cc, const float* __restrict__ scA,
    const float* __restrict__ scW, float* __restrict__ P,
    int M, int N, int K) {
    int gwave = (int)((blockIdx.x * blockDim.x + threadIdx.x) >> 5);
    int lane  = (int)(threadIdx.x & 31);
    int groupsN = N >> 6;
    int tm = gwave / groupsN, tg = gwave % groupsN;
    if (tm >= (M >> 4)) return;
    int hi = lane >> 4, lo = lane & 15;

    const int8_t* arow = A + (size_t)(tm * 16 + lo) * K + hi * 16;
    const int8_t* brow = B + (size_t)(tg * 64 + lo) * K + hi * 16;
    const size_t bstep = (size_t)16 * K;

    v8i acc0 = {0,0,0,0,0,0,0,0}, acc1 = {0,0,0,0,0,0,0,0};
    v8i acc2 = {0,0,0,0,0,0,0,0}, acc3 = {0,0,0,0,0,0,0,0};
    for (int k0 = 0; k0 < K; k0 += 64) {
        __builtin_prefetch(brow + k0 + 64, 0, 0);
        AV a;
#pragma unroll
        for (int c = 0; c < 4; ++c)
            a.d[c] = *(const long long*)(arow + k0 + (c >> 1) * 32 + (c & 1) * 8);
        BV b0, b1, b2, b3;
        b0.q[0] = *(const int4*)(brow + k0);
        b0.q[1] = *(const int4*)(brow + k0 + 32);
        b1.q[0] = *(const int4*)(brow + bstep + k0);
        b1.q[1] = *(const int4*)(brow + bstep + k0 + 32);
        b2.q[0] = *(const int4*)(brow + 2 * bstep + k0);
        b2.q[1] = *(const int4*)(brow + 2 * bstep + k0 + 32);
        b3.q[0] = *(const int4*)(brow + 3 * bstep + k0);
        b3.q[1] = *(const int4*)(brow + 3 * bstep + k0 + 32);
        acc0 = __builtin_amdgcn_wmma_i32_16x16x64_iu8(true, a.v, true, b0.v, acc0, false, false);
        acc1 = __builtin_amdgcn_wmma_i32_16x16x64_iu8(true, a.v, true, b1.v, acc1, false, false);
        acc2 = __builtin_amdgcn_wmma_i32_16x16x64_iu8(true, a.v, true, b2.v, acc2, false, false);
        acc3 = __builtin_amdgcn_wmma_i32_16x16x64_iu8(true, a.v, true, b3.v, acc3, false, false);
    }
    float SN2 = pow2_scale(*scA);
    float SN1 = pow2_scale(*scW);
#pragma unroll
    for (int j = 0; j < 4; ++j) {
        v8i acc = (j == 0) ? acc0 : (j == 1) ? acc1 : (j == 2) ? acc2 : acc3;
        int col = tg * 64 + j * 16 + lo;
        float cv = cc[col];
#pragma unroll
        for (int r = 0; r < 8; ++r) {
            int row = tm * 16 + r + hi * 8;
            P[(size_t)row * N + col] = (truncf((float)acc[r] / SN2) + cv) / SN1;
        }
    }
}


__global__ void colstats_kernel(const float* __restrict__ P,
                                float* __restrict__ mean,
                                float* __restrict__ var) {
    __shared__ float s1[256];
    __shared__ float s2[256];
    int col = blockIdx.x;
    float a = 0.0f, b = 0.0f;
    for (int r = threadIdx.x; r < M_; r += 256) {
        float v = P[(size_t)r * C_ + col];
        a += v; b += v * v;
    }
    s1[threadIdx.x] = a; s2[threadIdx.x] = b;
    __syncthreads();
    for (int s = 128; s > 0; s >>= 1) {
        if ((int)threadIdx.x < s) {
            s1[threadIdx.x] += s1[threadIdx.x + s];
            s2[threadIdx.x] += s2[threadIdx.x + s];
        }
        __syncthreads();
    }
    if (threadIdx.x == 0) {
        float mu = s1[0] / (float)M_;
        mean[col] = mu;
        var[col] = s2[0] / (float)M_ - mu * mu;
    }
}


__global__ void bn_lif_kernel(const float* __restrict__ P,
                              const float* __restrict__ mean,
                              const float* __restrict__ var,
                              const float* __restrict__ gamma,
                              const float* __restrict__ beta,
                              int do_bn, float vth, int mode, void* dst) {
    int idx = blockIdx.x * blockDim.x + threadIdx.x;
    if (idx >= B_ * NT_ * C_) return;
    int c  = idx % C_;
    int nt = (idx / C_) % NT_;
    int b  = idx / (C_ * NT_);
    float g = 1.0f, be = 0.0f, mu = 0.0f, inv = 1.0f;
    if (do_bn) {
        mu = mean[c]; inv = rsqrtf(var[c] + 1e-5f);
        g = gamma[c]; be = beta[c];
    }
    int h = c >> 5, dd = c & 31;
    float v = 0.0f;
    for (int t = 0; t < T_; ++t) {
        float x = P[(((size_t)t * B_ + b) * NT_ + nt) * C_ + c];
        if (do_bn) x = g * (x - mu) * inv + be;
        v += (x - v) * 0.5f;
        float sp = (v >= vth) ? 1.0f : 0.0f;
        v *= (1.0f - sp);
        size_t head = ((size_t)t * B_ + b) * H_ + h;
        if (mode == 0)
            ((int8_t*)dst)[head * (NPAD * 64) + (size_t)nt * 64 + dd] = (int8_t)sp;
        else if (mode == 1)
            ((int8_t*)dst)[head * (D_ * MPAD) + (size_t)dd * MPAD + nt] = (int8_t)sp;
        else
            ((float*)dst)[(((size_t)t * B_ + b) * NT_ + nt) * C_ + c] = sp;
    }
}


__global__ __launch_bounds__(256) void attn_kernel(
    const int8_t* __restrict__ qp, const int8_t* __restrict__ kp,
    const int8_t* __restrict__ vtp, float* __restrict__ y) {
    __shared__ __align__(16) int8_t S[NPAD * MPAD];
    int hid = blockIdx.x;
    const int8_t* q  = qp  + (size_t)hid * (NPAD * 64);
    const int8_t* k  = kp  + (size_t)hid * (NPAD * 64);
    const int8_t* vt = vtp + (size_t)hid * (D_ * MPAD);
    int wave = (int)(threadIdx.x >> 5), lane = (int)(threadIdx.x & 31);
    int hi = lane >> 4, lo = lane & 15;

    for (int i = (int)threadIdx.x * 16; i < NPAD * MPAD; i += 256 * 16) {
        int4 z = {0, 0, 0, 0};
        *(int4*)(&S[i]) = z;
    }
    __syncthreads();

    for (int tm = wave; tm < TN13; tm += 8) {
        AV a;
        const int8_t* ar = q + (size_t)(tm * 16 + lo) * 64 + hi * 16;
#pragma unroll
        for (int c = 0; c < 4; ++c)
            a.d[c] = *(const long long*)(ar + (c >> 1) * 32 + (c & 1) * 8);
        for (int tn = 0; tn < TN13; ++tn) {
            const int8_t* br = k + (size_t)(tn * 16 + lo) * 64 + hi * 16;
            __builtin_prefetch(br + 1024, 0, 0);
            BV b;
            b.q[0] = *(const int4*)(br);
            b.q[1] = *(const int4*)(br + 32);
            v8i acc = {0, 0, 0, 0, 0, 0, 0, 0};
            acc = __builtin_amdgcn_wmma_i32_16x16x64_iu8(false, a.v, false, b.v,
                                                         acc, false, false);
            int col = tn * 16 + lo;
#pragma unroll
            for (int r = 0; r < 8; ++r)
                S[(size_t)(tm * 16 + r + hi * 8) * MPAD + col] = (int8_t)acc[r];
        }
    }
    __syncthreads();

    int h = hid % H_, tb = hid / H_;
    for (int tm = wave; tm < TN13; tm += 8) {
        v8i acc0 = {0,0,0,0,0,0,0,0}, acc1 = {0,0,0,0,0,0,0,0};
        const int8_t* ar  = &S[(size_t)(tm * 16 + lo) * MPAD + hi * 16];
        const int8_t* br0 = vt + (size_t)lo * MPAD + hi * 16;
        const int8_t* br1 = vt + (size_t)(16 + lo) * MPAD + hi * 16;
        for (int k0 = 0; k0 < MPAD; k0 += 64) {
            AV a;
#pragma unroll
            for (int c = 0; c < 4; ++c)
                a.d[c] = *(const long long*)(ar + k0 + (c >> 1) * 32 + (c & 1) * 8);
            BV b0, b1;
            b0.q[0] = *(const int4*)(br0 + k0);
            b0.q[1] = *(const int4*)(br0 + k0 + 32);
            b1.q[0] = *(const int4*)(br1 + k0);
            b1.q[1] = *(const int4*)(br1 + k0 + 32);
            acc0 = __builtin_amdgcn_wmma_i32_16x16x64_iu8(false, a.v, false, b0.v,
                                                          acc0, false, false);
            acc1 = __builtin_amdgcn_wmma_i32_16x16x64_iu8(false, a.v, false, b1.v,
                                                          acc1, false, false);
        }
#pragma unroll
        for (int j = 0; j < 2; ++j) {
            v8i acc = j ? acc1 : acc0;
            int col = j * 16 + lo;
#pragma unroll
            for (int r = 0; r < 8; ++r) {
                int n = tm * 16 + r + hi * 8;
                if (n < NT_)
                    y[((size_t)tb * NT_ + n) * C_ + h * D_ + col] =
                        0.125f * (float)acc[r];
            }
        }
    }
}


extern "C" void kernel_launch(void* const* d_in, const int* in_sizes, int n_in,
                              void* d_out, int out_size, void* d_ws, size_t ws_size,
                              hipStream_t stream) {
    (void)in_sizes; (void)n_in; (void)out_size; (void)ws_size;
    const float* x = (const float*)d_in[0];

    char* ws = (char*)d_ws;
    int8_t* aa    = (int8_t*)(ws + 0);
    int8_t* bb    = (int8_t*)(ws + 4816896);
    float*  cc    = (float*)(ws + 4964352);
    float*  P     = (float*)(ws + 4965888);
    float*  mean  = (float*)(ws + 24233472);
    float*  var   = (float*)(ws + 24235008);
    int8_t* qpad  = (int8_t*)(ws + 24236544);
    int8_t* kpad  = (int8_t*)(ws + 34460160);
    int8_t* vtpad = (int8_t*)(ws + 44683776);
    float*  scal  = (float*)(ws + 50975232);

    const long long nX = (long long)M_ * C_;
    const int gemmBlocks = (M_ / 16) * (C_ / 64) / 8;
    const int elemBlocks = (B_ * NT_ * C_ + 255) / 256;
    const int wQBlocks = (C_ * C_ + 255) / 256;

    hipMemsetAsync(scal, 0, 8, stream);
    hipMemsetAsync(qpad, 0, (size_t)768 * NPAD * 64, stream);
    hipMemsetAsync(kpad, 0, (size_t)768 * NPAD * 64, stream);
    hipMemsetAsync(vtpad, 0, (size_t)768 * D_ * MPAD, stream);

    absmax_kernel<<<512, 256, 0, stream>>>(x, nX, &scal[0]);
    quant_i8_kernel<<<4704, 256, 0, stream>>>(x, &scal[0], aa, nX);

    for (int l = 0; l < 3; ++l) {
        const float* W    = (const float*)d_in[1 + 4 * l];
        const float* bias = (const float*)d_in[2 + 4 * l];
        const float* g    = (const float*)d_in[3 + 4 * l];
        const float* be   = (const float*)d_in[4 + 4 * l];
        hipMemsetAsync(&scal[1], 0, 4, stream);
        absmax_kernel<<<128, 256, 0, stream>>>(W, (long long)C_ * C_, &scal[1]);
        absmax_kernel<<<1, 256, 0, stream>>>(bias, (long long)C_, &scal[1]);
        quant_i8_kernel<<<wQBlocks, 256, 0, stream>>>(W, &scal[1], bb,
                                                      (long long)C_ * C_);
        quant_f32_kernel<<<2, 256, 0, stream>>>(bias, &scal[1], cc, C_);
        gemm_iu8_kernel<<<gemmBlocks, 256, 0, stream>>>(aa, bb, cc, &scal[0],
                                                        &scal[1], P, M_, C_, C_);
        colstats_kernel<<<C_, 256, 0, stream>>>(P, mean, var);
        void* dst = (l == 0) ? (void*)qpad : (l == 1) ? (void*)kpad : (void*)vtpad;
        int mode = (l == 2) ? 1 : 0;
        bn_lif_kernel<<<elemBlocks, 256, 0, stream>>>(P, mean, var, g, be, 1,
                                                      1.0f, mode, dst);
    }

    attn_kernel<<<T_ * B_ * H_, 256, 0, stream>>>(qpad, kpad, vtpad, P);
    bn_lif_kernel<<<elemBlocks, 256, 0, stream>>>(P, nullptr, nullptr, nullptr,
                                                  nullptr, 0, 0.5f, 2, (void*)P);

    const float* Wp = (const float*)d_in[13];
    const float* bp = (const float*)d_in[14];
    const float* gp = (const float*)d_in[15];
    const float* bep = (const float*)d_in[16];
    hipMemsetAsync(scal, 0, 8, stream);
    absmax_kernel<<<512, 256, 0, stream>>>(P, nX, &scal[0]);
    quant_i8_kernel<<<4704, 256, 0, stream>>>(P, &scal[0], aa, nX);
    absmax_kernel<<<128, 256, 0, stream>>>(Wp, (long long)C_ * C_, &scal[1]);
    absmax_kernel<<<1, 256, 0, stream>>>(bp, (long long)C_, &scal[1]);
    quant_i8_kernel<<<wQBlocks, 256, 0, stream>>>(Wp, &scal[1], bb,
                                                  (long long)C_ * C_);
    quant_f32_kernel<<<2, 256, 0, stream>>>(bp, &scal[1], cc, C_);
    gemm_iu8_kernel<<<gemmBlocks, 256, 0, stream>>>(aa, bb, cc, &scal[0],
                                                    &scal[1], P, M_, C_, C_);
    colstats_kernel<<<C_, 256, 0, stream>>>(P, mean, var);
    bn_lif_kernel<<<elemBlocks, 256, 0, stream>>>(P, mean, var, gp, bep, 1,
                                                  1.0f, 2, d_out);
}
